// BasicTransformerBlock_23210003268340
// MI455X (gfx1250) — matrix operand order repaired and checked by reading, not compiled, not run on hardware
//
#include <hip/hip_runtime.h>
#include <hip/hip_bf16.h>
#include <math.h>

// ---------------------------------------------------------------------------
// CDNA5 (gfx1250) wave32 WMMA implementation of the BasicTransformerBlock.
// bf16 operands staged into LDS with async-to-LDS DMA, f32 WMMA accumulation.
// ---------------------------------------------------------------------------

typedef __attribute__((ext_vector_type(16))) __bf16 v16bf;
typedef __attribute__((ext_vector_type(4)))  __bf16 v4bf;
typedef __attribute__((ext_vector_type(8)))  float  v8f;

#define B_DIM   2
#define S_DIM   2048
#define C_DIM   640
#define HEADS   8
#define DH      80       // main attention head dim
#define DHP     96       // DH padded to multiple of 32 for K-dim of WMMA
#define NTASKS  5
#define NATT    2
#define TDH     320      // task-attention head dim

__device__ __forceinline__ __bf16 f2bf(float f) { return (__bf16)f; }

// Async 16B/lane copy global->LDS (gfx1250). Generic LDS pointers carry the
// LDS byte offset in their low 32 bits (ISA 10.2 aperture scheme).
__device__ __forceinline__ void async_b128(void* lds, const void* gsrc) {
    unsigned off = (unsigned)(size_t)lds;
    asm volatile("global_load_async_to_lds_b128 %0, %1, off"
                 :: "v"(off), "v"(gsrc) : "memory");
}
__device__ __forceinline__ void wait_async0() {
    asm volatile("s_wait_asynccnt 0x0" ::: "memory");
}

// A fragment (16x32 MxK, bf16). Lane L(0-15): row M=L; elem e ->
// K = (L>>4)*16 + e.
__device__ __forceinline__ v16bf load_a_frag(const __bf16* base, int stride,
                                             int kofs, int lane) {
    const __bf16* p = base + (size_t)(lane & 15) * stride + kofs
                      + (((lane >> 4) & 1) << 4);
    v16bf a;
#pragma unroll
    for (int e = 0; e < 16; ++e)
        a[e] = p[e];
    return a;
}

// B fragment (32x16 KxN, bf16). Lane L: col N = L&15, K = (L>>4)*16 + e.
// 'base' is an [N][K]-layout LDS tile (row = N index, contiguous in K).
__device__ __forceinline__ v16bf load_b_frag(const __bf16* base, int stride,
                                             int kofs, int lane) {
    const __bf16* p = base + (size_t)(lane & 15) * stride + kofs
                      + (((lane >> 4) & 1) << 4);
    v16bf b;
#pragma unroll
    for (int e = 0; e < 16; ++e)
        b[e] = p[e];
    return b;
}

__device__ __forceinline__ v8f wmma_bf16(v16bf a, v16bf b, v8f c) {
    return __builtin_amdgcn_wmma_f32_16x16x32_bf16(false, a, false, b,
                                                   (short)0, c, false, false);
}

// ---------------------------------------------------------------------------
// f32 -> bf16 bulk convert (one-time per operand).
// ---------------------------------------------------------------------------
__global__ __launch_bounds__(256) void cvt_bf16_kernel(
    const float* __restrict__ x, __bf16* __restrict__ y, int n4)
{
    int i = blockIdx.x * 256 + threadIdx.x;
    if (i < n4) {
        float4 v = ((const float4*)x)[i];
        v4bf o;
        o[0] = f2bf(v.x); o[1] = f2bf(v.y); o[2] = f2bf(v.z); o[3] = f2bf(v.w);
        *(v4bf*)(y + 4 * (size_t)i) = o;
    }
}

// ---------------------------------------------------------------------------
// GEMM: C[M,N] = A[M,K] @ W[N,K]^T, A/W bf16, f32 accumulation.
// Block tile 128x128, 8 waves in 2(M)x4(N); each wave 64x32 -> 8 WMMA tiles.
// K stepped by 32; tiles staged by async-to-LDS DMA, double-buffered.
// Output modes: f32 (+bias/+resid), bf16, or bf16 per-head-transposed (for V).
// ---------------------------------------------------------------------------
#define GBM 128
#define GBN 128
#define GBK 32

__global__ __launch_bounds__(256) void gemm_bf16_kernel(
    const __bf16* __restrict__ A, const __bf16* __restrict__ W,
    const float* __restrict__ bias, const float* __restrict__ resid,
    float* __restrict__ outF, __bf16* __restrict__ outB,
    int M, int N, int K, int vt_mode)
{
    __shared__ __bf16 sA[2][GBM][GBK];   // 16 KB
    __shared__ __bf16 sW[2][GBN][GBK];   // 16 KB

    const int tid  = threadIdx.x;
    const int lane = tid & 31;
    const int wid  = tid >> 5;
    const int wm   = wid & 1;        // 2 waves along M (64 rows each)
    const int wn   = wid >> 1;       // 4 waves along N (32 cols each)
    const int m0   = blockIdx.y * GBM;
    const int n0   = blockIdx.x * GBN;

    v8f acc[4][2] = {};

    // 128x32 bf16 tile = 512 16B chunks; each thread DMAs 2 A + 2 W chunks.
    auto issue_tile = [&](int buf, int k0) {
#pragma unroll
        for (int i = 0; i < 2; ++i) {
            int cid = tid + i * 256;             // 0..511
            int r = cid >> 2, ce = (cid & 3) << 3;
            async_b128(&sA[buf][r][ce], A + (size_t)(m0 + r) * K + k0 + ce);
            async_b128(&sW[buf][r][ce], W + (size_t)(n0 + r) * K + k0 + ce);
        }
    };

    issue_tile(0, 0);
    wait_async0();
    __syncthreads();

    const int NIT = K / GBK;
    for (int it = 0; it < NIT; ++it) {
        const int cur = it & 1;
        if (it + 1 < NIT)
            issue_tile(cur ^ 1, (it + 1) * GBK);   // overlaps WMMA below

        v16bf aF[4], bF[2];
#pragma unroll
        for (int mt = 0; mt < 4; ++mt)
            aF[mt] = load_a_frag(&sA[cur][wm * 64 + mt * 16][0], GBK, 0, lane);
#pragma unroll
        for (int nt = 0; nt < 2; ++nt)
            bF[nt] = load_b_frag(&sW[cur][wn * 32 + nt * 16][0], GBK, 0, lane);
#pragma unroll
        for (int mt = 0; mt < 4; ++mt)
#pragma unroll
            for (int nt = 0; nt < 2; ++nt)
                acc[mt][nt] = wmma_bf16(aF[mt], bF[nt], acc[mt][nt]);

        if (it + 1 < NIT)
            wait_async0();
        __syncthreads();
    }

    // C/D layout: elem r -> (M = (lane>>4)*8 + r, N = lane&15)
    const int mh = (lane >> 4) & 1, nl = lane & 15;
#pragma unroll
    for (int mt = 0; mt < 4; ++mt)
#pragma unroll
        for (int nt = 0; nt < 2; ++nt)
#pragma unroll
            for (int r = 0; r < 8; ++r) {
                int m = m0 + wm * 64 + mt * 16 + mh * 8 + r;
                int n = n0 + wn * 32 + nt * 16 + nl;
                float val = acc[mt][nt][r];
                if (vt_mode) {
                    // V projection: scatter to [b, h, d, s] bf16
                    int hh = n / DH, d = n - hh * DH;
                    int bb = m >> 11, s = m & (S_DIM - 1);
                    outB[(((size_t)bb * HEADS + hh) * DH + d) * S_DIM + s] =
                        f2bf(val);
                } else if (outB) {
                    outB[(size_t)m * N + n] = f2bf(val);
                } else {
                    if (bias)  val += bias[n];
                    if (resid) val += resid[(size_t)m * N + n];
                    outF[(size_t)m * N + n] = val;
                }
            }
}

// ---------------------------------------------------------------------------
// Flash attention: heads=8, dh=80 (padded to 96). Q/K bf16 [B,S,640],
// V bf16 pre-transposed [B,H,DH,S]. Softmax scale folded into scores.
// One block per (128 query rows, b*h); 8 waves; KV in 32-row blocks with
// double-buffered async-to-LDS staging; online softmax; WMMA matmuls.
// ---------------------------------------------------------------------------
__global__ __launch_bounds__(256) void flash_attn_kernel(
    const __bf16* __restrict__ Qbf, const __bf16* __restrict__ Kbf,
    const __bf16* __restrict__ Vt, __bf16* __restrict__ Og)
{
    __shared__ __bf16 sQ[128][DHP];         // 24 KB
    __shared__ __bf16 sK[2][32][DHP];       // 12 KB
    __shared__ __bf16 sVt[2][DH][32];       // 10 KB ([d][kv])
    __shared__ __bf16 sP[8][16][32];        //  8 KB (per-wave P tile)

    const int tid  = threadIdx.x;
    const int lane = tid & 31;
    const int wid  = tid >> 5;
    const int m0   = blockIdx.x * 128;
    const int bh   = blockIdx.y;
    const int b    = bh / HEADS, h = bh % HEADS;
    const size_t rowbase = (size_t)b * S_DIM;
    const __bf16* Vth = Vt + (size_t)bh * DH * S_DIM;
    const float qscale = 0.11180339887498949f;   // 1/sqrt(80)

    // Zero the pad columns (d=80..95) once; DMA only touches d<80.
    for (int idx = tid; idx < 128 * 16; idx += 256)
        sQ[idx >> 4][DH + (idx & 15)] = f2bf(0.0f);
    for (int idx = tid; idx < 2 * 32 * 16; idx += 256)
        sK[idx >> 9][(idx >> 4) & 31][DH + (idx & 15)] = f2bf(0.0f);

    // Stage Q: 128 rows x 80 bf16 = 1280 16B chunks, 5 per thread (async).
#pragma unroll
    for (int i = 0; i < 5; ++i) {
        int cid = tid + i * 256;
        int r = cid / 10, ce = (cid % 10) * 8;
        async_b128(&sQ[r][ce],
                   Qbf + (rowbase + m0 + r) * C_DIM + h * DH + ce);
    }

    // Stage one KV block: K 320 chunks + Vt 320 chunks (async).
    auto issue_kv = [&](int buf, int kv0) {
#pragma unroll
        for (int i = 0; i < 3; ++i) {
            int v = tid + i * 256;
            if (v < 320) {
                int r = v / 10, ce = (v % 10) * 8;
                async_b128(&sK[buf][r][ce],
                           Kbf + (rowbase + kv0 + r) * C_DIM + h * DH + ce);
            } else if (v < 640) {
                int u = v - 320;
                int d = u >> 2, ce = (u & 3) << 3;
                async_b128(&sVt[buf][d][ce],
                           Vth + (size_t)d * S_DIM + kv0 + ce);
            }
        }
    };

    issue_kv(0, 0);
    wait_async0();
    __syncthreads();

    // Persistent per-wave Q fragments, K-dim chunks 0..95.
    v16bf qF[3];
#pragma unroll
    for (int c = 0; c < 3; ++c)
        qF[c] = load_a_frag(&sQ[wid * 16][0], DHP, c * 32, lane);

    float mrow[8], lrow[8];
#pragma unroll
    for (int r = 0; r < 8; ++r) { mrow[r] = -3.0e38f; lrow[r] = 0.0f; }
    v8f oacc[5] = {};

    const int mh = (lane >> 4) & 1, nl = lane & 15;
    const int NKV = S_DIM / 32;

    for (int it = 0; it < NKV; ++it) {
        const int cur = it & 1;
        if (it + 1 < NKV)
            issue_kv(cur ^ 1, (it + 1) * 32);   // overlaps compute below

        // Scores S = Q K^T : 2 kv-column tiles x 3 K-chunks of WMMA.
        v8f sc[2] = {};
#pragma unroll
        for (int t = 0; t < 2; ++t)
#pragma unroll
            for (int c = 0; c < 3; ++c) {
                v16bf bK = load_b_frag(&sK[cur][t * 16][0], DHP, c * 32, lane);
                sc[t] = wmma_bf16(qF[c], bK, sc[t]);
            }

        // Online softmax (scale folded in). Row m = mh*8+r spans the 16 lanes
        // of a half; xor masks 1..8 flip only lane bits 0..3.
        float scale[8];
#pragma unroll
        for (int r = 0; r < 8; ++r) {
            float s0 = sc[0][r] * qscale, s1 = sc[1][r] * qscale;
            float v = fmaxf(s0, s1);
#pragma unroll
            for (int o = 1; o < 16; o <<= 1)
                v = fmaxf(v, __shfl_xor(v, o));
            float nm = fmaxf(mrow[r], v);
            float sr = __expf(mrow[r] - nm);
            mrow[r] = nm;
            float p0 = __expf(s0 - nm);
            float p1 = __expf(s1 - nm);
            float ps = p0 + p1;
#pragma unroll
            for (int o = 1; o < 16; o <<= 1)
                ps += __shfl_xor(ps, o);
            lrow[r] = lrow[r] * sr + ps;
            scale[r] = sr;
            sP[wid][mh * 8 + r][nl]      = f2bf(p0);
            sP[wid][mh * 8 + r][16 + nl] = f2bf(p1);
        }
#pragma unroll
        for (int nt = 0; nt < 5; ++nt)
#pragma unroll
            for (int r = 0; r < 8; ++r)
                oacc[nt][r] *= scale[r];

        // O += P V : 5 dh-tiles of WMMA (same-wave LDS ops are in order).
        v16bf pF = load_a_frag(&sP[wid][0][0], 32, 0, lane);
#pragma unroll
        for (int nt = 0; nt < 5; ++nt) {
            v16bf bV = load_b_frag(&sVt[cur][nt * 16][0], 32, 0, lane);
            oacc[nt] = wmma_bf16(pF, bV, oacc[nt]);
        }

        if (it + 1 < NKV)
            wait_async0();
        __syncthreads();
    }

    float inv[8];
#pragma unroll
    for (int r = 0; r < 8; ++r) inv[r] = 1.0f / lrow[r];
#pragma unroll
    for (int nt = 0; nt < 5; ++nt)
#pragma unroll
        for (int r = 0; r < 8; ++r) {
            int gr = m0 + wid * 16 + mh * 8 + r;
            Og[(rowbase + gr) * C_DIM + h * DH + nt * 16 + nl] =
                f2bf(oacc[nt][r] * inv[r]);
        }
}

// ---------------------------------------------------------------------------
// LayerNorm over C=640, one row per block (8 waves); bf16 output.
// ---------------------------------------------------------------------------
__global__ __launch_bounds__(256) void layernorm_kernel(
    const float* __restrict__ x, const float* __restrict__ g,
    const float* __restrict__ bta, __bf16* __restrict__ y)
{
    const int row  = blockIdx.x;
    const int tid  = threadIdx.x;
    const int lane = tid & 31;
    const int wid  = tid >> 5;
    const float* xr = x + (size_t)row * C_DIM;

    float s = 0.0f, ss = 0.0f;
    for (int e = tid; e < C_DIM; e += 256) {
        float v = xr[e]; s += v; ss += v * v;
    }
#pragma unroll
    for (int o = 16; o >= 1; o >>= 1) { s += __shfl_xor(s, o); ss += __shfl_xor(ss, o); }

    __shared__ float ps[8], pss[8], stat[2];
    if (lane == 0) { ps[wid] = s; pss[wid] = ss; }
    __syncthreads();
    if (tid == 0) {
        float ts = 0.0f, tss = 0.0f;
        for (int i = 0; i < 8; ++i) { ts += ps[i]; tss += pss[i]; }
        float mean = ts / (float)C_DIM;
        float var  = tss / (float)C_DIM - mean * mean;
        stat[0] = mean;
        stat[1] = rsqrtf(var + 1e-5f);
    }
    __syncthreads();
    float mean = stat[0], rstd = stat[1];
    __bf16* yr = y + (size_t)row * C_DIM;
    for (int e = tid; e < C_DIM; e += 256)
        yr[e] = f2bf((xr[e] - mean) * rstd * g[e] + bta[e]);
}

// ---------------------------------------------------------------------------
// Task attention: per (b,s,head2) a length-1 query over the 5-task axis,
// dh=320 -> one wave per unit; f32 in, bf16 out (feeds the final GEMM).
// ---------------------------------------------------------------------------
__global__ __launch_bounds__(256) void task_attn_kernel(
    const float* __restrict__ tq, const float* __restrict__ tk,
    const float* __restrict__ tv, __bf16* __restrict__ outp)
{
    const int tid  = threadIdx.x;
    const int lane = tid & 31;
    const int wid  = tid >> 5;
    const int unit = blockIdx.x * 8 + wid;       // 0..B*S*NATT-1
    const int bs   = unit >> 1;
    const int h    = unit & 1;
    const size_t qb = (size_t)bs * C_DIM + (size_t)h * TDH;
    const float tscale = 0.05590169943749474f;   // 1/sqrt(320)

    float qv[10];
#pragma unroll
    for (int i = 0; i < 10; ++i) qv[i] = tq[qb + lane + 32 * i];

    float w[NTASKS];
    float mx = -3.0e38f;
#pragma unroll
    for (int t = 0; t < NTASKS; ++t) {
        const size_t kb = ((size_t)t * (B_DIM * S_DIM) + bs) * C_DIM + (size_t)h * TDH;
        float d = 0.0f;
#pragma unroll
        for (int i = 0; i < 10; ++i) d += qv[i] * tk[kb + lane + 32 * i];
#pragma unroll
        for (int o = 16; o >= 1; o >>= 1) d += __shfl_xor(d, o);
        w[t] = d * tscale;
        mx = fmaxf(mx, w[t]);
    }
    float den = 0.0f;
#pragma unroll
    for (int t = 0; t < NTASKS; ++t) { w[t] = __expf(w[t] - mx); den += w[t]; }
    float invd = 1.0f / den;

    float o[10] = {};
#pragma unroll
    for (int t = 0; t < NTASKS; ++t) {
        const size_t vb = ((size_t)t * (B_DIM * S_DIM) + bs) * C_DIM + (size_t)h * TDH;
        float wt = w[t] * invd;
#pragma unroll
        for (int i = 0; i < 10; ++i) o[i] += wt * tv[vb + lane + 32 * i];
    }
#pragma unroll
    for (int i = 0; i < 10; ++i) outp[qb + lane + 32 * i] = f2bf(o[i]);
}

// ---------------------------------------------------------------------------
extern "C" void kernel_launch(void* const* d_in, const int* in_sizes, int n_in,
                              void* d_out, int out_size, void* d_ws, size_t ws_size,
                              hipStream_t stream) {
    (void)in_sizes; (void)n_in; (void)out_size; (void)ws_size;

    const float* hs   = (const float*)d_in[0];
    const float* tf   = (const float*)d_in[1];
    const float* Wq   = (const float*)d_in[2];
    const float* Wk   = (const float*)d_in[3];
    const float* Wv   = (const float*)d_in[4];
    const float* Wo   = (const float*)d_in[5];
    const float* bo   = (const float*)d_in[6];
    const float* lnqg = (const float*)d_in[7];
    const float* lnqb = (const float*)d_in[8];
    const float* Wtq  = (const float*)d_in[9];
    const float* lnkg = (const float*)d_in[10];
    const float* lnkb = (const float*)d_in[11];
    const float* Wtk  = (const float*)d_in[12];
    const float* lnvg = (const float*)d_in[13];
    const float* lnvb = (const float*)d_in[14];
    const float* Wtv  = (const float*)d_in[15];
    const float* Wot  = (const float*)d_in[16];
    const float* bot  = (const float*)d_in[17];
    float* out = (float*)d_out;

    const size_t NM = (size_t)B_DIM * S_DIM * C_DIM;   // 2,621,440
    const size_t WN = (size_t)C_DIM * C_DIM;           //   409,600

    // f32 region
    float* ws   = (float*)d_ws;
    float* hs1  = ws;                 // [B*S,C] f32
    float* tqb  = ws + 1 * NM;        // [B*S,C] f32
    float* tkb  = ws + 2 * NM;        // [T*B*S,C] f32 (5*NM)
    float* tvb  = ws + 7 * NM;        // [T*B*S,C] f32 (5*NM)
    // bf16 region
    __bf16* bfb   = (__bf16*)(ws + 12 * NM);
    __bf16* hs_bf = bfb;              // [B*S,C]
    __bf16* q_bf  = bfb + 1 * NM;
    __bf16* k_bf  = bfb + 2 * NM;
    __bf16* vt_bf = bfb + 3 * NM;     // [B,H,DH,S]
    __bf16* at_bf = bfb + 4 * NM;     // attention out
    __bf16* lq_bf = bfb + 5 * NM;     // LN(hs1)
    __bf16* lk_bf = bfb + 6 * NM;     // LN(task_feat), 5*NM, reused
    __bf16* to_bf = bfb + 11 * NM;    // task-attn out
    __bf16* w_bf  = bfb + 12 * NM;    // 8 weight matrices
    __bf16* Wq_b  = w_bf + 0 * WN;
    __bf16* Wk_b  = w_bf + 1 * WN;
    __bf16* Wv_b  = w_bf + 2 * WN;
    __bf16* Wo_b  = w_bf + 3 * WN;
    __bf16* Wtq_b = w_bf + 4 * WN;
    __bf16* Wtk_b = w_bf + 5 * WN;
    __bf16* Wtv_b = w_bf + 6 * WN;
    __bf16* Wot_b = w_bf + 7 * WN;

    const int M1 = B_DIM * S_DIM;        // 4096
    const int MT = NTASKS * M1;          // 20480

    dim3 blk(256);
    dim3 g1(C_DIM / GBN, M1 / GBM);      // (5, 32)
    dim3 gT(C_DIM / GBN, MT / GBM);      // (5, 160)
    const int cvA = (int)(NM / 4 / 256); // 2560 blocks
    const int cvW = (int)(WN / 4 / 256); //  400 blocks

    // One-time bf16 conversions (weights + hidden_states).
    cvt_bf16_kernel<<<dim3(cvA), blk, 0, stream>>>(hs,  hs_bf, (int)(NM / 4));
    cvt_bf16_kernel<<<dim3(cvW), blk, 0, stream>>>(Wq,  Wq_b,  (int)(WN / 4));
    cvt_bf16_kernel<<<dim3(cvW), blk, 0, stream>>>(Wk,  Wk_b,  (int)(WN / 4));
    cvt_bf16_kernel<<<dim3(cvW), blk, 0, stream>>>(Wv,  Wv_b,  (int)(WN / 4));
    cvt_bf16_kernel<<<dim3(cvW), blk, 0, stream>>>(Wo,  Wo_b,  (int)(WN / 4));
    cvt_bf16_kernel<<<dim3(cvW), blk, 0, stream>>>(Wtq, Wtq_b, (int)(WN / 4));
    cvt_bf16_kernel<<<dim3(cvW), blk, 0, stream>>>(Wtk, Wtk_b, (int)(WN / 4));
    cvt_bf16_kernel<<<dim3(cvW), blk, 0, stream>>>(Wtv, Wtv_b, (int)(WN / 4));
    cvt_bf16_kernel<<<dim3(cvW), blk, 0, stream>>>(Wot, Wot_b, (int)(WN / 4));

    // q, k (bf16 out), v (bf16 transposed-per-head out)
    gemm_bf16_kernel<<<g1, blk, 0, stream>>>(hs_bf, Wq_b, nullptr, nullptr,
                                             nullptr, q_bf, M1, C_DIM, C_DIM, 0);
    gemm_bf16_kernel<<<g1, blk, 0, stream>>>(hs_bf, Wk_b, nullptr, nullptr,
                                             nullptr, k_bf, M1, C_DIM, C_DIM, 0);
    gemm_bf16_kernel<<<g1, blk, 0, stream>>>(hs_bf, Wv_b, nullptr, nullptr,
                                             nullptr, vt_bf, M1, C_DIM, C_DIM, 1);

    // self attention
    dim3 gFA(S_DIM / 128, B_DIM * HEADS);  // (16, 16)
    flash_attn_kernel<<<gFA, blk, 0, stream>>>(q_bf, k_bf, vt_bf, at_bf);

    // hs1 = attn @ Wo^T + bo (f32)
    gemm_bf16_kernel<<<g1, blk, 0, stream>>>(at_bf, Wo_b, bo, nullptr,
                                             hs1, nullptr, M1, C_DIM, C_DIM, 0);

    // task query path
    layernorm_kernel<<<dim3(M1), blk, 0, stream>>>(hs1, lnqg, lnqb, lq_bf);
    gemm_bf16_kernel<<<g1, blk, 0, stream>>>(lq_bf, Wtq_b, nullptr, nullptr,
                                             tqb, nullptr, M1, C_DIM, C_DIM, 0);

    // task keys / values (LN buffer reused sequentially)
    layernorm_kernel<<<dim3(MT), blk, 0, stream>>>(tf, lnkg, lnkb, lk_bf);
    gemm_bf16_kernel<<<gT, blk, 0, stream>>>(lk_bf, Wtk_b, nullptr, nullptr,
                                             tkb, nullptr, MT, C_DIM, C_DIM, 0);
    layernorm_kernel<<<dim3(MT), blk, 0, stream>>>(tf, lnvg, lnvb, lk_bf);
    gemm_bf16_kernel<<<gT, blk, 0, stream>>>(lk_bf, Wtv_b, nullptr, nullptr,
                                             tvb, nullptr, MT, C_DIM, C_DIM, 0);

    // 5-way task attention (bf16 out)
    task_attn_kernel<<<dim3(M1 * NATT / 8), blk, 0, stream>>>(tqb, tkb, tvb, to_bf);

    // out = tout @ Wot^T + bot + hs1 (f32)
    gemm_bf16_kernel<<<g1, blk, 0, stream>>>(to_bf, Wot_b, bot, hs1,
                                             out, nullptr, M1, C_DIM, C_DIM, 0);
}
